// GeometryAwareCrossAttention_15522011807843
// MI455X (gfx1250) — matrix operand order repaired and checked by reading, not compiled, not run on hardware
//
#include <hip/hip_runtime.h>
#include <hip/hip_bf16.h>

// ---------------------------------------------------------------------------
// Types for CDNA5 WMMA
// ---------------------------------------------------------------------------
typedef __attribute__((ext_vector_type(16))) __bf16 v16bf;
typedef __attribute__((ext_vector_type(8)))  float  v8f;

union BF16Frag {
    v16bf v;
    unsigned short u[16];
    uint4 q[2];
};

__device__ __forceinline__ unsigned short f2bf(float f) {
    unsigned int x = __float_as_uint(f);
    unsigned int r = x + 0x7FFFu + ((x >> 16) & 1u);   // round-to-nearest-even
    return (unsigned short)(r >> 16);
}

__device__ __forceinline__ float wave_sum(float v) {
    #pragma unroll
    for (int m = 16; m >= 1; m >>= 1) v += __shfl_xor(v, m, 32);
    return v;
}
__device__ __forceinline__ float wave_max(float v) {
    #pragma unroll
    for (int m = 16; m >= 1; m >>= 1) v = fmaxf(v, __shfl_xor(v, m, 32));
    return v;
}

// ---------------------------------------------------------------------------
// 1. block ranges from sorted block_id
// ---------------------------------------------------------------------------
__global__ void ranges_kernel(const int* __restrict__ bid, int* __restrict__ bs,
                              int N, int NB) {
    int i = blockIdx.x * 256 + threadIdx.x;
    if (i >= N) return;
    int id   = bid[i];
    int prev = (i == 0) ? -1 : bid[i - 1];
    for (int b = prev + 1; b <= id; ++b) bs[b] = i;
    if (i == N - 1)
        for (int b = id + 1; b <= NB; ++b) bs[b] = N;
}

// ---------------------------------------------------------------------------
// 2. centroids: one wave per block
// ---------------------------------------------------------------------------
__global__ void centroid_kernel(const float* __restrict__ pos,
                                const int* __restrict__ bs,
                                float* __restrict__ cent, int NB) {
    int b = blockIdx.x * 8 + (threadIdx.x >> 5);
    if (b >= NB) return;
    int lane = threadIdx.x & 31;
    int s = bs[b], e = bs[b + 1];
    float sx = 0.f, sy = 0.f, sz = 0.f;
    for (int i = s + lane; i < e; i += 32) {
        sx += pos[3 * i + 0];
        sy += pos[3 * i + 1];
        sz += pos[3 * i + 2];
    }
    sx = wave_sum(sx); sy = wave_sum(sy); sz = wave_sum(sz);
    if (lane == 0) {
        float c = fmaxf((float)(e - s), 1.0f);
        cent[3 * b + 0] = sx / c;
        cent[3 * b + 1] = sy / c;
        cent[3 * b + 2] = sz / c;
    }
}

// ---------------------------------------------------------------------------
// 3. weight convert + transpose: W[K][Nc] f32 row-major -> WT[Nc][K] bf16
// ---------------------------------------------------------------------------
__global__ void convw_kernel(const float* __restrict__ W,
                             unsigned short* __restrict__ WT,
                             int Kdim, int Ncol) {
    int t = blockIdx.x * 256 + threadIdx.x;
    if (t >= Kdim * Ncol) return;
    int kk = t / Ncol, n = t % Ncol;
    WT[(size_t)n * Kdim + kk] = f2bf(W[t]);
}

// ---------------------------------------------------------------------------
// 4. fused RBF + geom MLP + bf16 aug row (one wave per atom)
// ---------------------------------------------------------------------------
__global__ void geom_kernel(const float* __restrict__ af,
                            const float* __restrict__ pos,
                            const int* __restrict__ bid,
                            const float* __restrict__ cent,
                            const float* __restrict__ rc,
                            const float* __restrict__ rw,
                            const float* __restrict__ Wg,
                            const float* __restrict__ bg,
                            unsigned short* __restrict__ aug, int N) {
    int i = blockIdx.x * 8 + (threadIdx.x >> 5);
    if (i >= N) return;
    int lane = threadIdx.x & 31;
    int sg = bid[i];
    float dx = pos[3 * i + 0] - cent[3 * sg + 0];
    float dy = pos[3 * i + 1] - cent[3 * sg + 1];
    float dz = pos[3 * i + 2] - cent[3 * sg + 2];
    float d = sqrtf(dx * dx + dy * dy + dz * dz);
    float r[16];
    #pragma unroll
    for (int j = 0; j < 16; ++j) {
        float df = d - rc[j];
        float w  = rw[j];
        r[j] = __expf(-(df * df) / (2.0f * w * w));
    }
    unsigned short* arow = aug + (size_t)i * 320;
    const float*    frow = af  + (size_t)i * 256;
    #pragma unroll
    for (int j = 0; j < 8; ++j) {
        int c = lane + 32 * j;
        arow[c] = f2bf(frow[c]);
    }
    #pragma unroll
    for (int rep = 0; rep < 2; ++rep) {
        int g = lane + 32 * rep;
        float acc = bg[g];
        #pragma unroll
        for (int j = 0; j < 16; ++j) acc += r[j] * Wg[j * 64 + g];
        arow[256 + g] = f2bf(acc);
    }
}

// ---------------------------------------------------------------------------
// 5. Generic WMMA GEMM: C[M,Nc] = act( A[M,K] @ W[K,Nc] + bias )
//    WT is bf16 column-major [Nc][K]. Each wave: 16x64 strip (4 WMMA tiles).
// ---------------------------------------------------------------------------
template<int ABF16, int RELU, int OUTBF16>
__global__ void gemm_wmma_kernel(const void* __restrict__ Aptr, int lda,
                                 const unsigned short* __restrict__ WT, int Kdim,
                                 const float* __restrict__ bias,
                                 void* __restrict__ Cptr, int ldc,
                                 int M, int Ncols) {
    int lane = threadIdx.x & 31;
    int wave = threadIdx.x >> 5;
    int l15  = lane & 15;
    int half = lane >> 4;                 // 0 or 1
    int n0 = (blockIdx.x * 4 + wave) * 64;
    if (n0 >= Ncols) return;
    int m0 = blockIdx.y * 16;
    int ra = m0 + l15; if (ra >= M) ra = M - 1;
    int kbA = half * 8;                   // A: K offset per §7.12.2
    int kbB = half * 8;                   // B: K offset per §7.12.2

    const unsigned short* wrow[4];
    #pragma unroll
    for (int t = 0; t < 4; ++t)
        wrow[t] = WT + (size_t)(n0 + t * 16 + l15) * Kdim;

    v8f acc[4] = {};
    if (ABF16) {
        const unsigned short* arow = (const unsigned short*)Aptr + (size_t)ra * lda;
        for (int k = 0; k < Kdim; k += 32) {
            BF16Frag a;
            a.q[0] = *(const uint4*)(arow + k + kbA);
            a.q[1] = *(const uint4*)(arow + k + 16 + kbA);
            #pragma unroll
            for (int t = 0; t < 4; ++t) {
                BF16Frag b;
                b.q[0] = *(const uint4*)(wrow[t] + k + kbB);
                b.q[1] = *(const uint4*)(wrow[t] + k + 16 + kbB);
                acc[t] = __builtin_amdgcn_wmma_f32_16x16x32_bf16(
                    false, a.v, false, b.v, (short)0, acc[t], false, false);
            }
        }
    } else {
        const float* arow = (const float*)Aptr + (size_t)ra * lda;
        for (int k = 0; k < Kdim; k += 32) {
            BF16Frag a;
            float4 f0 = *(const float4*)(arow + k + kbA);
            float4 f1 = *(const float4*)(arow + k + kbA + 4);
            float4 f2 = *(const float4*)(arow + k + 16 + kbA);
            float4 f3 = *(const float4*)(arow + k + 16 + kbA + 4);
            a.u[0] = f2bf(f0.x);  a.u[1] = f2bf(f0.y);
            a.u[2] = f2bf(f0.z);  a.u[3] = f2bf(f0.w);
            a.u[4] = f2bf(f1.x);  a.u[5] = f2bf(f1.y);
            a.u[6] = f2bf(f1.z);  a.u[7] = f2bf(f1.w);
            a.u[8] = f2bf(f2.x);  a.u[9] = f2bf(f2.y);
            a.u[10] = f2bf(f2.z); a.u[11] = f2bf(f2.w);
            a.u[12] = f2bf(f3.x); a.u[13] = f2bf(f3.y);
            a.u[14] = f2bf(f3.z); a.u[15] = f2bf(f3.w);
            #pragma unroll
            for (int t = 0; t < 4; ++t) {
                BF16Frag b;
                b.q[0] = *(const uint4*)(wrow[t] + k + kbB);
                b.q[1] = *(const uint4*)(wrow[t] + k + 16 + kbB);
                acc[t] = __builtin_amdgcn_wmma_f32_16x16x32_bf16(
                    false, a.v, false, b.v, (short)0, acc[t], false, false);
            }
        }
    }

    #pragma unroll
    for (int t = 0; t < 4; ++t) {
        int cb = n0 + t * 16 + l15;
        float bv = bias ? bias[cb] : 0.0f;
        #pragma unroll
        for (int r = 0; r < 8; ++r) {
            int rowr = m0 + half * 8 + r;       // D layout §7.12.2
            if (rowr >= M) continue;
            float v = acc[t][r] + bv;
            if (RELU) v = fmaxf(v, 0.0f);
            if (OUTBF16)
                ((unsigned short*)Cptr)[(size_t)rowr * ldc + cb] = f2bf(v);
            else
                ((float*)Cptr)[(size_t)rowr * ldc + cb] = v;
        }
    }
}

// ---------------------------------------------------------------------------
// 6. scores[i] = dot(Q[seg[i]], K[i]) / 16  (one wave per atom)
// ---------------------------------------------------------------------------
__global__ void scores_kernel(const float* __restrict__ Q,
                              const float* __restrict__ Kb,
                              const int* __restrict__ bid,
                              float* __restrict__ sc, int N) {
    int i = blockIdx.x * 8 + (threadIdx.x >> 5);
    if (i >= N) return;
    int lane = threadIdx.x & 31;
    const float4* q4 = (const float4*)(Q  + (size_t)bid[i] * 256);
    const float4* k4 = (const float4*)(Kb + (size_t)i * 256);
    float4 a0 = q4[lane], a1 = q4[lane + 32];
    float4 b0 = k4[lane], b1 = k4[lane + 32];
    float p = a0.x * b0.x + a0.y * b0.y + a0.z * b0.z + a0.w * b0.w
            + a1.x * b1.x + a1.y * b1.y + a1.z * b1.z + a1.w * b1.w;
    p = wave_sum(p);
    if (lane == 0) sc[i] = p * 0.0625f;   // 1/sqrt(256)
}

// ---------------------------------------------------------------------------
// 7. segment softmax + context (one 256-thread workgroup per block)
// ---------------------------------------------------------------------------
__global__ void ctx_kernel(const float* __restrict__ sc,
                           const float* __restrict__ V,
                           const int* __restrict__ bs,
                           float* __restrict__ ctx, int NB) {
    __shared__ float red[256];
    __shared__ float wsh[256];
    __shared__ float s_m, s_d;
    int b = blockIdx.x;
    int t = threadIdx.x;
    int s = bs[b], e = bs[b + 1];

    float mx = -3.4e38f;
    for (int i = s + t; i < e; i += 256) mx = fmaxf(mx, sc[i]);
    red[t] = mx; __syncthreads();
    for (int off = 128; off > 0; off >>= 1) {
        if (t < off) red[t] = fmaxf(red[t], red[t + off]);
        __syncthreads();
    }
    if (t == 0) s_m = (red[0] > -3.0e38f) ? red[0] : 0.0f;
    __syncthreads();
    float m = s_m;

    float ds = 0.f;
    for (int i = s + t; i < e; i += 256) ds += __expf(sc[i] - m);
    red[t] = ds; __syncthreads();
    for (int off = 128; off > 0; off >>= 1) {
        if (t < off) red[t] += red[t + off];
        __syncthreads();
    }
    if (t == 0) s_d = fmaxf(red[0], 1e-20f);
    __syncthreads();
    float invd = 1.0f / s_d;

    float acc = 0.f;
    for (int base = s; base < e; base += 256) {
        int cnt = e - base; if (cnt > 256) cnt = 256;
        __syncthreads();
        if (t < cnt) wsh[t] = __expf(sc[base + t] - m) * invd;
        __syncthreads();
        for (int j = 0; j < cnt; ++j)
            acc += wsh[j] * V[(size_t)(base + j) * 256 + t];
    }
    ctx[(size_t)b * 256 + t] = acc;
}

// ---------------------------------------------------------------------------
// 8. LayerNorm(x + y) — wave per atom. SEG=1: y gathered per block id.
// ---------------------------------------------------------------------------
template<int SEG>
__global__ void ln_kernel(const float* __restrict__ X,
                          const float* __restrict__ Y,
                          const int* __restrict__ bid,
                          const float* __restrict__ g,
                          const float* __restrict__ bb,
                          float* __restrict__ out, int N) {
    int i = blockIdx.x * 8 + (threadIdx.x >> 5);
    if (i >= N) return;
    int lane = threadIdx.x & 31;
    const float4* x4 = (const float4*)(X + (size_t)i * 256);
    const float4* y4 = SEG ? (const float4*)(Y + (size_t)bid[i] * 256)
                           : (const float4*)(Y + (size_t)i * 256);
    float4 v0 = x4[lane],    u0 = y4[lane];
    float4 v1 = x4[lane+32], u1 = y4[lane+32];
    v0.x += u0.x; v0.y += u0.y; v0.z += u0.z; v0.w += u0.w;
    v1.x += u1.x; v1.y += u1.y; v1.z += u1.z; v1.w += u1.w;
    float s  = v0.x + v0.y + v0.z + v0.w + v1.x + v1.y + v1.z + v1.w;
    float ss = v0.x*v0.x + v0.y*v0.y + v0.z*v0.z + v0.w*v0.w
             + v1.x*v1.x + v1.y*v1.y + v1.z*v1.z + v1.w*v1.w;
    s  = wave_sum(s);
    ss = wave_sum(ss);
    float mean = s * (1.0f / 256.0f);
    float var  = ss * (1.0f / 256.0f) - mean * mean;
    float inv  = rsqrtf(var + 1e-5f);
    const float4* g4 = (const float4*)g;
    const float4* b4 = (const float4*)bb;
    float4 G0 = g4[lane], B0 = b4[lane], G1 = g4[lane+32], B1 = b4[lane+32];
    float4 o0, o1;
    o0.x = (v0.x - mean) * inv * G0.x + B0.x;
    o0.y = (v0.y - mean) * inv * G0.y + B0.y;
    o0.z = (v0.z - mean) * inv * G0.z + B0.z;
    o0.w = (v0.w - mean) * inv * G0.w + B0.w;
    o1.x = (v1.x - mean) * inv * G1.x + B1.x;
    o1.y = (v1.y - mean) * inv * G1.y + B1.y;
    o1.z = (v1.z - mean) * inv * G1.z + B1.z;
    o1.w = (v1.w - mean) * inv * G1.w + B1.w;
    float4* o4 = (float4*)(out + (size_t)i * 256);
    o4[lane] = o0; o4[lane + 32] = o1;
}

// ---------------------------------------------------------------------------
// Launch
// ---------------------------------------------------------------------------
static inline size_t align256(size_t x) { return (x + 255) & ~(size_t)255; }

extern "C" void kernel_launch(void* const* d_in, const int* in_sizes, int n_in,
                              void* d_out, int out_size, void* d_ws, size_t ws_size,
                              hipStream_t stream) {
    const float* af   = (const float*)d_in[0];
    const float* pos  = (const float*)d_in[1];
    const float* bfeat= (const float*)d_in[2];
    const int*   bid  = (const int*)  d_in[3];
    const float* rbfc = (const float*)d_in[4];
    const float* rbfw = (const float*)d_in[5];
    const float* Wg   = (const float*)d_in[6];
    const float* bg   = (const float*)d_in[7];
    const float* Wq   = (const float*)d_in[8];
    const float* bq   = (const float*)d_in[9];
    const float* Wk   = (const float*)d_in[10];
    const float* bk   = (const float*)d_in[11];
    const float* Wv   = (const float*)d_in[12];
    const float* bv   = (const float*)d_in[13];
    const float* Wc1  = (const float*)d_in[14];
    const float* bc1  = (const float*)d_in[15];
    const float* Wc2  = (const float*)d_in[16];
    const float* bc2  = (const float*)d_in[17];
    const float* Wf1  = (const float*)d_in[18];
    const float* bf1  = (const float*)d_in[19];
    const float* Wf2  = (const float*)d_in[20];
    const float* bf2  = (const float*)d_in[21];
    const float* ln1g = (const float*)d_in[22];
    const float* ln1b = (const float*)d_in[23];
    const float* ln2g = (const float*)d_in[24];
    const float* ln2b = (const float*)d_in[25];

    const int N  = in_sizes[1] / 3;      // 500000
    const int NB = in_sizes[2] / 256;    // 50000

    // ---- workspace layout ----
    char* cur = (char*)d_ws;
    auto alloc = [&](size_t bytes) { char* p = cur; cur += align256(bytes); return p; };
    int*            bs_ptr  = (int*)           alloc((size_t)(NB + 1) * 4);
    float*          cent    = (float*)         alloc((size_t)NB * 3 * 4);
    float*          Qbuf    = (float*)         alloc((size_t)NB * 256 * 4); // then t1
    float*          ctxbuf  = (float*)         alloc((size_t)NB * 256 * 4);
    float*          updbuf  = (float*)         alloc((size_t)NB * 256 * 4);
    float*          scbuf   = (float*)         alloc((size_t)N * 4);
    unsigned short* WqT     = (unsigned short*)alloc((size_t)256 * 256 * 2);
    unsigned short* WkT     = (unsigned short*)alloc((size_t)256 * 320 * 2);
    unsigned short* WvT     = (unsigned short*)alloc((size_t)256 * 320 * 2);
    unsigned short* Wc1T    = (unsigned short*)alloc((size_t)256 * 256 * 2);
    unsigned short* Wc2T    = (unsigned short*)alloc((size_t)256 * 256 * 2);
    unsigned short* Wf1T    = (unsigned short*)alloc((size_t)512 * 256 * 2);
    unsigned short* Wf2T    = (unsigned short*)alloc((size_t)256 * 512 * 2);
    unsigned short* aug     = (unsigned short*)alloc((size_t)N * 320 * 2);
    float*          buf1    = (float*)         alloc((size_t)N * 256 * 4); // K, then ffn2
    float*          buf2    = (float*)         alloc((size_t)N * 256 * 4); // V, then h
    unsigned short* ffn1    = (unsigned short*)alloc((size_t)N * 512 * 2);
    (void)ws_size; (void)n_in; (void)out_size;

    dim3 b256(256);
    dim3 b128(128);
    auto ggrid = [](int M, int Nc) { return dim3((Nc + 255) / 256, (M + 15) / 16); };
    int wavesN  = (N + 7) / 8;
    int wavesNB = (NB + 7) / 8;

    // 1. ranges + centroids
    ranges_kernel<<<(N + 255) / 256, b256, 0, stream>>>(bid, bs_ptr, N, NB);
    centroid_kernel<<<wavesNB, b256, 0, stream>>>(pos, bs_ptr, cent, NB);

    // 2. weight conversions (bf16, column-major)
    convw_kernel<<<(256 * 256 + 255) / 256, b256, 0, stream>>>(Wq,  WqT,  256, 256);
    convw_kernel<<<(320 * 256 + 255) / 256, b256, 0, stream>>>(Wk,  WkT,  320, 256);
    convw_kernel<<<(320 * 256 + 255) / 256, b256, 0, stream>>>(Wv,  WvT,  320, 256);
    convw_kernel<<<(256 * 256 + 255) / 256, b256, 0, stream>>>(Wc1, Wc1T, 256, 256);
    convw_kernel<<<(256 * 256 + 255) / 256, b256, 0, stream>>>(Wc2, Wc2T, 256, 256);
    convw_kernel<<<(256 * 512 + 255) / 256, b256, 0, stream>>>(Wf1, Wf1T, 256, 512);
    convw_kernel<<<(512 * 256 + 255) / 256, b256, 0, stream>>>(Wf2, Wf2T, 512, 256);

    // 3. RBF + geom MLP -> aug (bf16 N x 320)
    geom_kernel<<<wavesN, b256, 0, stream>>>(af, pos, bid, cent, rbfc, rbfw,
                                             Wg, bg, aug, N);

    // 4. Q = block_features @ Wq + bq
    gemm_wmma_kernel<0,0,0><<<ggrid(NB,256), b128, 0, stream>>>(
        bfeat, 256, WqT, 256, bq, Qbuf, 256, NB, 256);
    // 5. K = aug @ Wk + bk ; V = aug @ Wv + bv
    gemm_wmma_kernel<1,0,0><<<ggrid(N,256), b128, 0, stream>>>(
        aug, 320, WkT, 320, bk, buf1, 256, N, 256);
    gemm_wmma_kernel<1,0,0><<<ggrid(N,256), b128, 0, stream>>>(
        aug, 320, WvT, 320, bv, buf2, 256, N, 256);

    // 6. scores, segment softmax + context
    scores_kernel<<<wavesN, b256, 0, stream>>>(Qbuf, buf1, bid, scbuf, N);
    ctx_kernel<<<NB, b256, 0, stream>>>(scbuf, buf2, bs_ptr, ctxbuf, NB);

    // 7. upd = relu(ctx@Wc1+bc1)@Wc2+bc2   (Qbuf recycled as hidden)
    gemm_wmma_kernel<0,1,0><<<ggrid(NB,256), b128, 0, stream>>>(
        ctxbuf, 256, Wc1T, 256, bc1, Qbuf, 256, NB, 256);
    gemm_wmma_kernel<0,0,0><<<ggrid(NB,256), b128, 0, stream>>>(
        Qbuf, 256, Wc2T, 256, bc2, updbuf, 256, NB, 256);

    // 8. h = LN1(atom_features + upd[seg]) -> buf2 (V dead)
    ln_kernel<1><<<wavesN, b256, 0, stream>>>(af, updbuf, bid, ln1g, ln1b, buf2, N);

    // 9. FFN: ffn1 = relu(h@Wf1+bf1) bf16 ; ffn2 = ffn1@Wf2+bf2 -> buf1 (K dead)
    gemm_wmma_kernel<0,1,1><<<ggrid(N,512), b128, 0, stream>>>(
        buf2, 256, Wf1T, 256, bf1, ffn1, 512, N, 512);
    gemm_wmma_kernel<1,0,0><<<ggrid(N,256), b128, 0, stream>>>(
        ffn1, 512, Wf2T, 512, bf2, buf1, 256, N, 256);

    // 10. out = LN2(h + ffn)
    ln_kernel<0><<<wavesN, b256, 0, stream>>>(buf2, buf1, (const int*)nullptr,
                                              ln2g, ln2b, (float*)d_out, N);
}
